// ManualMultiHeadSelfAttHead_12567074308581
// MI455X (gfx1250) — matrix operand order repaired and checked by reading, not compiled, not run on hardware
//
#include <hip/hip_runtime.h>

// ---------------------------------------------------------------------------
// MI455X / gfx1250 implementation, bf16 WMMA + double-buffered async LDS loads.
// scores = X*Wqk*X^T ; column softmax (over q) ; out = P * (X*Wvk^T)
// ---------------------------------------------------------------------------

typedef __attribute__((ext_vector_type(16))) __bf16 v16bf;
typedef __attribute__((ext_vector_type(8)))  float  v8f;
typedef unsigned short u16;

#define Bdim 16
#define Sdim 512
#define Ddim 768
#define Hdim 16
#define Odim 48

union ABFrag { uint4 q[2]; v16bf v; };

__device__ __forceinline__ u16 f2bf(float f) {
    union { float f; unsigned u; } c; c.f = f;
    unsigned r = c.u + 0x7FFFu + ((c.u >> 16) & 1u);
    return (u16)(r >> 16);
}

__device__ __forceinline__ unsigned lds_off(const void* p) {
    return (unsigned)(unsigned long long)p;   // low 32 bits == LDS offset
}

// CDNA5 async global->LDS copy (ASYNCcnt); 16 bytes per active lane.
__device__ __forceinline__ void async_ld_b128(unsigned ldsAddr, const void* gaddr) {
    asm volatile("global_load_async_to_lds_b128 %0, %1, off"
                 :: "v"(ldsAddr), "v"(gaddr) : "memory");
}
// Wait until per-wave ASYNCcnt <= N (immediate).
#define WAIT_ASYNC(n) asm volatile("s_wait_asynccnt %0" :: "i"(n) : "memory")

__device__ __forceinline__ v8f wmma_bf16(const ABFrag& a, const ABFrag& b, v8f c) {
    return __builtin_amdgcn_wmma_f32_16x16x32_bf16(
        /*neg_a=*/false, a.v, /*neg_b=*/false, b.v,
        /*c_mod=*/(short)0, c, /*reuse_a=*/false, /*reuse_b=*/false);
}

// ---------------------------------------------------------------------------
// fp32 -> bf16 conversion
// ---------------------------------------------------------------------------
__global__ __launch_bounds__(256) void f32_to_bf16_kernel(const float* __restrict__ in,
                                                          u16* __restrict__ out, int n) {
    int i = blockIdx.x * 256 + threadIdx.x;
    if (i < n) out[i] = f2bf(in[i]);
}

// ---------------------------------------------------------------------------
// Tiled bf16 GEMM: C[M,N] = A[M,K] * B  (B row-major KxN, or TRANS_B: NxK)
// 256 threads = 8 waves (2x4), block tile 64x128, wave tile 32x32, Ktile 64,
// double-buffered LDS with overlapped async global->LDS copies.
// blockIdx.z = head index; per-head strides in elements.
// Requires: M%64==0, K%64==0, N%16==0 (zero-padded in LDS when N < tile).
// ---------------------------------------------------------------------------
template<bool TRANS_B, bool OUT_BF16>
__global__ __launch_bounds__(256) void gemm_bf16_kernel(
        const u16* __restrict__ Abase, const u16* __restrict__ Bbase, void* Cbase,
        int M, int N, int K, int lda, int ldb, int ldc,
        int strideA, int strideB, int strideC)
{
    __shared__ u16 ldsA[2][64][64];    // [buf][m][k]
    __shared__ u16 ldsBt[2][128][64];  // [buf][n][k]  N-major: B frags contiguous in K

    const int z    = blockIdx.z;
    const u16* A   = Abase + (size_t)z * strideA;
    const u16* B   = Bbase + (size_t)z * strideB;
    const int tid  = threadIdx.x;
    const int wave = tid >> 5;
    const int lane = tid & 31;
    const int wm   = wave >> 2;          // 0..1
    const int wn   = wave & 3;           // 0..3
    const int blockM = blockIdx.y * 64;
    const int blockN = blockIdx.x * 128;

    const unsigned offA = lds_off(&ldsA[0][0][0]);
    const unsigned offB = lds_off(&ldsBt[0][0][0]);

    // ---- tile fill helpers (per-thread) -----------------------------------
    const int am = tid >> 2, akc = tid & 3;            // A: 64 rows x 4 k-chunks
    const int bn = tid >> 1, bkh = tid & 1;            // B(T): 128 rows x 2 k-halves
    const int bk = tid >> 2, bng = tid & 3;            // B(N): 64 k x 4 n-groups
    const bool bInB = TRANS_B ? ((blockN + bn) < N) : false;

    auto fillA = [&](int kk, int buf) {
        const u16* src = A + (size_t)(blockM + am) * lda + kk + akc * 8;
        unsigned dst = offA + (unsigned)buf * (64 * 64 * 2)
                            + (unsigned)(am * 64 + akc * 8) * 2;
        async_ld_b128(dst,      src);        // k: [akc*8 .. +7]
        async_ld_b128(dst + 64, src + 32);   // k: [32+akc*8 .. +7]
        if (kk + 64 < K) __builtin_prefetch(src + 64);
    };
    auto fillB = [&](int kk, int buf) {
        if constexpr (TRANS_B) {
            unsigned dst = offB + (unsigned)buf * (128 * 64 * 2)
                                + (unsigned)(bn * 64 + bkh * 32) * 2;
            if (bInB) {
                const u16* src = B + (size_t)(blockN + bn) * ldb + kk + bkh * 32;
                async_ld_b128(dst,      src);
                async_ld_b128(dst + 16, src + 8);
                async_ld_b128(dst + 32, src + 16);
                async_ld_b128(dst + 48, src + 24);
            } else {
                uint4 zero = make_uint4(0u, 0u, 0u, 0u);
                uint4* zp = (uint4*)&ldsBt[buf][bn][bkh * 32];
                zp[0] = zero; zp[1] = zero; zp[2] = zero; zp[3] = zero;
            }
        } else {
            // transpose while filling: src row-major [k][n]
            int gN = blockN + bng * 32;
            const u16* src = B + (size_t)(kk + bk) * ldb + gN;
#pragma unroll
            for (int e = 0; e < 32; ++e) {
                u16 val = (gN + e < N) ? src[e] : (u16)0;
                ldsBt[buf][bng * 32 + e][bk] = val;
            }
        }
    };

    v8f acc[2][2] = {};
    const int NT = K >> 6;               // K / 64 tiles

    fillA(0, 0);
    fillB(0, 0);

    for (int t = 0; t < NT; ++t) {
        const int buf = t & 1;
        if (t + 1 < NT) {
            fillA((t + 1) << 6, buf ^ 1);
            fillB((t + 1) << 6, buf ^ 1);
            // wait for *this* tile's copies only (in-order completion):
            // outstanding for next tile = 2 (A) [+4 if in-bounds TRANS_B].
            if (TRANS_B && bInB) { WAIT_ASYNC(6); } else { WAIT_ASYNC(2); }
        } else {
            WAIT_ASYNC(0);
        }
        __syncthreads();

        // ---- compute: 2 K-steps x (2x2) WMMA tiles per wave
#pragma unroll
        for (int ks = 0; ks < 2; ++ks) {
            const int kb = ks * 32;
            ABFrag a0, a1;
            {
                int r0 = wm * 32 + (lane & 15);
                int r1 = r0 + 16;
                int ab = kb + (lane >> 4) * 8;   // A layout: K {0..7,16..23}/{8..15,24..31}
                a0.q[0] = *(const uint4*)&ldsA[buf][r0][ab];
                a0.q[1] = *(const uint4*)&ldsA[buf][r0][ab + 16];
                a1.q[0] = *(const uint4*)&ldsA[buf][r1][ab];
                a1.q[1] = *(const uint4*)&ldsA[buf][r1][ab + 16];
            }
#pragma unroll
            for (int ni = 0; ni < 2; ++ni) {
                ABFrag bf;
                int br = wn * 32 + ni * 16 + (lane & 15);
                int bb = kb + (lane >> 4) * 8;   // B layout: K {0..7,16..23}/{8..15,24..31}
                bf.q[0] = *(const uint4*)&ldsBt[buf][br][bb];
                bf.q[1] = *(const uint4*)&ldsBt[buf][br][bb + 16];
                acc[0][ni] = wmma_bf16(a0, bf, acc[0][ni]);
                acc[1][ni] = wmma_bf16(a1, bf, acc[1][ni]);
            }
        }
        __syncthreads();
    }

    // ---- epilogue: C layout VGPR i -> M = i + 8*(lane>=16), N = lane&15
    const int lm = (lane >> 4) << 3;
    const int ln = lane & 15;
#pragma unroll
    for (int mi = 0; mi < 2; ++mi)
#pragma unroll
        for (int ni = 0; ni < 2; ++ni)
#pragma unroll
            for (int i = 0; i < 8; ++i) {
                int gm = blockM + wm * 32 + mi * 16 + i + lm;
                int gn = blockN + wn * 32 + ni * 16 + ln;
                if (gm < M && gn < N) {
                    size_t off = (size_t)z * strideC + (size_t)gm * ldc + gn;
                    if constexpr (OUT_BF16) ((u16*)Cbase)[off]   = f2bf(acc[mi][ni][i]);
                    else                    ((float*)Cbase)[off] = acc[mi][ni][i];
                }
            }
}

// ---------------------------------------------------------------------------
// Column softmax stats over q (axis=2 of the reference): per (h,k)
// ---------------------------------------------------------------------------
__global__ __launch_bounds__(256) void colstats_kernel(const float* __restrict__ S,
                                                       float* __restrict__ colmax,
                                                       float* __restrict__ colsum) {
    int idx = blockIdx.x * 256 + threadIdx.x;     // h*512 + k
    int h = idx >> 9, k = idx & 511;
    const float* col = S + (size_t)h * Sdim * Sdim + k;
    float mx = -3.0e38f;
    for (int q = 0; q < Sdim; ++q) mx = fmaxf(mx, col[(size_t)q * Sdim]);
    float sm = 0.f;
    for (int q = 0; q < Sdim; ++q) sm += __expf(col[(size_t)q * Sdim] - mx);
    colmax[idx] = mx;
    colsum[idx] = sm;
}

// ---------------------------------------------------------------------------
// out[b,q,h*48+o] = sum_k exp(S[h,q,k]-colmax[k])/colsum[k] * V[h,k,o]
// Block 256 = 8 waves, each wave: 16 q rows x 48 outputs (3 WMMA n-tiles).
// grid = (S/128, H)
// ---------------------------------------------------------------------------
__global__ __launch_bounds__(256) void attn_out_kernel(
        const float* __restrict__ S, const float* __restrict__ colmax,
        const float* __restrict__ colsum, const u16* __restrict__ Vbf,
        float* __restrict__ out)
{
    __shared__ u16 ldsP[128][32];   // probability tile, bf16, A layout
    __shared__ u16 ldsVt[Odim][32]; // V tile transposed: [o][k]

    const int tid  = threadIdx.x;
    const int wave = tid >> 5;
    const int lane = tid & 31;
    const int qbase = blockIdx.x * 128;
    const int h     = blockIdx.y;

    const float* Sh  = S + (size_t)h * Sdim * Sdim;
    const float* cmx = colmax + h * Sdim;
    const float* csm = colsum + h * Sdim;
    const u16*   Vh  = Vbf + (size_t)h * Sdim * Odim;

    v8f acc[3] = {};

    for (int ks = 0; ks < Sdim; ks += 32) {
        // build P tile in bf16 (16 elems per thread)
        {
            int m = tid >> 1, kh = tid & 1;
            const float* srow = Sh + (size_t)(qbase + m) * Sdim + ks + kh * 16;
            u16* dst = &ldsP[m][kh * 16];
#pragma unroll
            for (int e = 0; e < 16; ++e) {
                int k = ks + kh * 16 + e;
                float p = __expf(srow[e] - cmx[k]) / csm[k];
                dst[e] = f2bf(p);
            }
        }
        // stage V tile transposed
        if (tid < 2 * Odim) {
            int o = tid >> 1, kh = tid & 1;
#pragma unroll
            for (int e = 0; e < 16; ++e) {
                int k = ks + kh * 16 + e;
                ldsVt[o][kh * 16 + e] = Vh[(size_t)k * Odim + o];
            }
        }
        __syncthreads();

        ABFrag a;
        {
            int ar = wave * 16 + (lane & 15);
            int ab = (lane >> 4) * 8;
            a.q[0] = *(const uint4*)&ldsP[ar][ab];
            a.q[1] = *(const uint4*)&ldsP[ar][ab + 16];
        }
#pragma unroll
        for (int nt = 0; nt < 3; ++nt) {
            ABFrag bf;
            int br = nt * 16 + (lane & 15);
            int bb = (lane >> 4) * 8;
            bf.q[0] = *(const uint4*)&ldsVt[br][bb];
            bf.q[1] = *(const uint4*)&ldsVt[br][bb + 16];
            acc[nt] = wmma_bf16(a, bf, acc[nt]);
        }
        __syncthreads();
    }

    const int lm = (lane >> 4) << 3;
    const int ln = lane & 15;
#pragma unroll
    for (int nt = 0; nt < 3; ++nt) {
        int o = h * Odim + nt * 16 + ln;
#pragma unroll
        for (int i = 0; i < 8; ++i) {
            int gq = qbase + wave * 16 + i + lm;
            out[(size_t)gq * Ddim + o] = acc[nt][i];
        }
    }
}

// ---------------------------------------------------------------------------
// Host-side launch: workspace layout (bytes, all 256-aligned), ~71 MB total
// ---------------------------------------------------------------------------
extern "C" void kernel_launch(void* const* d_in, const int* in_sizes, int n_in,
                              void* d_out, int out_size, void* d_ws, size_t ws_size,
                              hipStream_t stream) {
    (void)in_sizes; (void)n_in; (void)out_size; (void)ws_size;

    const float* e_s = (const float*)d_in[0];   // [B,S,D]
    const float* Wqk = (const float*)d_in[1];   // [H,D,D]
    const float* Wvk = (const float*)d_in[2];   // [H,O,D]
    float* out = (float*)d_out;                 // [B,S,H*O]

    char* ws = (char*)d_ws;
    const size_t SZ_XBF = (size_t)Bdim * Sdim * Ddim * 2;
    const size_t SZ_WQK = (size_t)Hdim * Ddim * Ddim * 2;
    const size_t SZ_WVK = (size_t)Hdim * Odim * Ddim * 2;
    const size_t SZ_V   = (size_t)Bdim * Hdim * Sdim * Odim * 2;
    const size_t SZ_M   = (size_t)Hdim * Sdim * Ddim * 2;
    const size_t SZ_S   = (size_t)Hdim * Sdim * Sdim * 4;
    const size_t SZ_CM  = (size_t)Hdim * Sdim * 4;

    u16*   Xbf   = (u16*)(ws);
    u16*   Wqkb  = (u16*)(ws + SZ_XBF);
    u16*   Wvkb  = (u16*)(ws + SZ_XBF + SZ_WQK);
    u16*   Vbf   = (u16*)(ws + SZ_XBF + SZ_WQK + SZ_WVK);
    u16*   Mbf   = (u16*)(ws + SZ_XBF + SZ_WQK + SZ_WVK + SZ_V);
    float* Sbuf  = (float*)(ws + SZ_XBF + SZ_WQK + SZ_WVK + SZ_V + SZ_M);
    float* cmax  = (float*)(ws + SZ_XBF + SZ_WQK + SZ_WVK + SZ_V + SZ_M + SZ_S);
    float* csum  = (float*)(ws + SZ_XBF + SZ_WQK + SZ_WVK + SZ_V + SZ_M + SZ_S + SZ_CM);

    const int nX   = Bdim * Sdim * Ddim;
    const int nWqk = Hdim * Ddim * Ddim;
    const int nWvk = Hdim * Odim * Ddim;
    f32_to_bf16_kernel<<<(nX   + 255) / 256, 256, 0, stream>>>(e_s, Xbf, nX);
    f32_to_bf16_kernel<<<(nWqk + 255) / 256, 256, 0, stream>>>(Wqk, Wqkb, nWqk);
    f32_to_bf16_kernel<<<(nWvk + 255) / 256, 256, 0, stream>>>(Wvk, Wvkb, nWvk);

    for (int b = 0; b < Bdim; ++b) {
        const u16* Xb = Xbf + (size_t)b * Sdim * Ddim;
        u16* Vb = Vbf + (size_t)b * Hdim * Sdim * Odim;

        // M[h] = X[b] @ Wqk[h]         (512 x 768 x 768), bf16 out
        gemm_bf16_kernel<false, true><<<dim3(Ddim / 128, Sdim / 64, Hdim), 256, 0, stream>>>(
            Xb, Wqkb, Mbf, Sdim, Ddim, Ddim, Ddim, Ddim, Ddim,
            0, Ddim * Ddim, Sdim * Ddim);

        // V[h] = X[b] @ Wvk[h]^T       (512 x 48 x 768), bf16 out
        gemm_bf16_kernel<true, true><<<dim3(1, Sdim / 64, Hdim), 256, 0, stream>>>(
            Xb, Wvkb, Vb, Sdim, Odim, Ddim, Ddim, Ddim, Odim,
            0, Odim * Ddim, Sdim * Odim);

        // S[h] = M[h] @ X[b]^T         (512 x 512 x 768), fp32 out
        gemm_bf16_kernel<true, false><<<dim3(Sdim / 128, Sdim / 64, Hdim), 256, 0, stream>>>(
            Mbf, Xb, Sbuf, Sdim, Sdim, Ddim, Ddim, Ddim, Sdim,
            Sdim * Ddim, 0, Sdim * Sdim);

        // column softmax stats (over q)
        colstats_kernel<<<Hdim * Sdim / 256, 256, 0, stream>>>(Sbuf, cmax, csum);

        // out[b] = P @ V
        attn_out_kernel<<<dim3(Sdim / 128, Hdim), 256, 0, stream>>>(
            Sbuf, cmax, csum, Vb, out + (size_t)b * Sdim * Ddim);
    }
}
